// PotentialPredictor_9268539424845
// MI455X (gfx1250) — matrix operand order repaired and checked by reading, not compiled, not run on hardware
//
#include <hip/hip_runtime.h>
#include <hip/hip_bf16.h>
#include <math.h>

typedef __attribute__((ext_vector_type(16))) __bf16 v16bf;
typedef __attribute__((ext_vector_type(8)))  __bf16 v8bf;
typedef __attribute__((ext_vector_type(8)))  float  v8f;

#define NNODES   20000
#define NEDGES   320000
#define NGRAPHS  32
#define CDIM     256
#define EDIM     518   // 2*C + 6
#define LN_EPS   1e-5f

__device__ __forceinline__ float gelu_exact(float v) {
    return 0.5f * v * (1.0f + erff(v * 0.70710678118654752f));
}

// -------------------------------------------------------------------------
// Pack f32 [M,K] -> bf16 [M,Kp], zero padding cols K..Kp-1.
// -------------------------------------------------------------------------
__global__ void pack_cast_pad(const float* __restrict__ in, __bf16* __restrict__ out,
                              int M, int K, int Kp) {
    int idx = blockIdx.x * blockDim.x + threadIdx.x;
    if (idx >= M * Kp) return;
    int row = idx / Kp, k = idx - row * Kp;
    float v = (k < K) ? in[(size_t)row * K + k] : 0.0f;
    out[idx] = (__bf16)v;
}

// Pack + transpose: W f32 [K,N] -> WT bf16 [N,Kp], zero padding k >= K.
__global__ void pack_transpose(const float* __restrict__ W, __bf16* __restrict__ WT,
                               int K, int N, int Kp) {
    int idx = blockIdx.x * blockDim.x + threadIdx.x;
    if (idx >= N * Kp) return;
    int n = idx / Kp, k = idx - n * Kp;
    float v = (k < K) ? W[(size_t)k * N + n] : 0.0f;
    WT[idx] = (__bf16)v;
}

// -------------------------------------------------------------------------
// WMMA GEMM: OUT = act(A @ W + bias), A: bf16 [M,Kp] row-major,
// WT: bf16 [N,Kp] (i.e. W transposed), bias[N] f32.
// Block = 256 threads = 8 waves. Each wave: 16-row strip x 4 col tiles.
// Grid: (ceil(M/128), N/64). Requires M % 16 == 0, Kp % 32 == 0.
// Inner loop: branch-free b128 loads + 4 WMMAs.
// -------------------------------------------------------------------------
__global__ void gemm_wmma_kernel(const __bf16* __restrict__ A,
                                 const __bf16* __restrict__ WT,
                                 const float* __restrict__ bias,
                                 float* __restrict__ OUT32,
                                 __bf16* __restrict__ OUT16,
                                 int M, int N, int Kp, int act) {
    const int lane = threadIdx.x & 31;
    const int wave = threadIdx.x >> 5;
    const int g    = lane >> 4;      // lane group (0 or 1)
    const int ln   = lane & 15;
    const int m0   = blockIdx.x * 128 + wave * 16;
    const int n0   = blockIdx.y * 64;
    if (m0 >= M) return;             // wave-uniform: EXEC stays all-1s

    // A fragment base for this lane: row m0+ln, halves offset 16*g
    const __bf16* aRow = A + (size_t)(m0 + ln) * Kp + 16 * g;
    // B fragment bases: column n0+nt*16+ln of W == row of WT, offset 16*g
    const __bf16* bCol0 = WT + (size_t)(n0 +  0 + ln) * Kp + 16 * g;
    const __bf16* bCol1 = WT + (size_t)(n0 + 16 + ln) * Kp + 16 * g;
    const __bf16* bCol2 = WT + (size_t)(n0 + 32 + ln) * Kp + 16 * g;
    const __bf16* bCol3 = WT + (size_t)(n0 + 48 + ln) * Kp + 16 * g;

    v8f acc0 = {}, acc1 = {}, acc2 = {}, acc3 = {};

    for (int k0 = 0; k0 < Kp; k0 += 32) {
        // A 16x32 bf16 fragment (k = t + 16g):
        // halves [k0+16g, +16) contiguous  -> two b128 loads
        v8bf alo = *(const v8bf*)(aRow + k0);
        v8bf ahi = *(const v8bf*)(aRow + k0 + 8);
        v16bf a = __builtin_shufflevector(alo, ahi,
                    0,1,2,3,4,5,6,7,8,9,10,11,12,13,14,15);
        // B 32x16 fragment (k = t + 16g): 16 contiguous halves -> two b128 loads
        {
            v8bf blo = *(const v8bf*)(bCol0 + k0);
            v8bf bhi = *(const v8bf*)(bCol0 + k0 + 8);
            v16bf b = __builtin_shufflevector(blo, bhi,
                        0,1,2,3,4,5,6,7,8,9,10,11,12,13,14,15);
            acc0 = __builtin_amdgcn_wmma_f32_16x16x32_bf16(
                false, a, false, b, (short)0, acc0, false, false);
        }
        {
            v8bf blo = *(const v8bf*)(bCol1 + k0);
            v8bf bhi = *(const v8bf*)(bCol1 + k0 + 8);
            v16bf b = __builtin_shufflevector(blo, bhi,
                        0,1,2,3,4,5,6,7,8,9,10,11,12,13,14,15);
            acc1 = __builtin_amdgcn_wmma_f32_16x16x32_bf16(
                false, a, false, b, (short)0, acc1, false, false);
        }
        {
            v8bf blo = *(const v8bf*)(bCol2 + k0);
            v8bf bhi = *(const v8bf*)(bCol2 + k0 + 8);
            v16bf b = __builtin_shufflevector(blo, bhi,
                        0,1,2,3,4,5,6,7,8,9,10,11,12,13,14,15);
            acc2 = __builtin_amdgcn_wmma_f32_16x16x32_bf16(
                false, a, false, b, (short)0, acc2, false, false);
        }
        {
            v8bf blo = *(const v8bf*)(bCol3 + k0);
            v8bf bhi = *(const v8bf*)(bCol3 + k0 + 8);
            v16bf b = __builtin_shufflevector(blo, bhi,
                        0,1,2,3,4,5,6,7,8,9,10,11,12,13,14,15);
            acc3 = __builtin_amdgcn_wmma_f32_16x16x32_bf16(
                false, a, false, b, (short)0, acc3, false, false);
        }
    }

    // Epilogue: bias + optional GELU; C/D layout: m = m0 + r + 8*g, n = lane%16
#pragma unroll
    for (int nt = 0; nt < 4; ++nt) {
        int col = n0 + nt * 16 + ln;
        float bv = bias[col];
        v8f accv = (nt == 0) ? acc0 : (nt == 1) ? acc1 : (nt == 2) ? acc2 : acc3;
#pragma unroll
        for (int r = 0; r < 8; ++r) {
            int m = m0 + r + 8 * g;
            float v = accv[r] + bv;
            if (act) v = gelu_exact(v);
            if (OUT32) OUT32[(size_t)m * N + col] = v;
            if (OUT16) OUT16[(size_t)m * N + col] = (__bf16)v;
        }
    }
}

// -------------------------------------------------------------------------
// Prep per layer: gw1[i][k] = ln1_g[i]*w1_w[i][k]; econst[k]=sum_i g*w1,
// econst[4+k]=sum_i b*w1. Single block.
// -------------------------------------------------------------------------
__global__ void prep_kernel(const float* __restrict__ ln1_g,
                            const float* __restrict__ ln1_b,
                            const float* __restrict__ w1_w,
                            float* __restrict__ gw1,
                            float* __restrict__ econst) {
    __shared__ float acc[8];
    if (threadIdx.x < 8) acc[threadIdx.x] = 0.0f;
    __syncthreads();
    float loc[8] = {0, 0, 0, 0, 0, 0, 0, 0};
    for (int i = threadIdx.x; i < EDIM; i += blockDim.x) {
        float gg = ln1_g[i], bb = ln1_b[i];
#pragma unroll
        for (int k = 0; k < 4; ++k) {
            float w = w1_w[i * 4 + k];
            gw1[i * 4 + k] = gg * w;
            loc[k]     += gg * w;
            loc[4 + k] += bb * w;
        }
    }
#pragma unroll
    for (int k = 0; k < 8; ++k) atomicAdd(&acc[k], loc[k]);
    __syncthreads();
    if (threadIdx.x < 8) econst[threadIdx.x] = acc[threadIdx.x];
}

__global__ void zero_kernel(float* __restrict__ p, int n) {
    int i = blockIdx.x * blockDim.x + threadIdx.x;
    if (i < n) p[i] = 0.0f;
}

// -------------------------------------------------------------------------
// Edge kernel: one wave32 per edge. LN1+[518->4] folded into 6 wave
// reductions; gate computed redundantly per lane; f32 atomic scatter.
// -------------------------------------------------------------------------
__global__ void edge_kernel(const float* __restrict__ mfeat,
                            const float* __restrict__ x_pos,
                            const int* __restrict__ src,
                            const int* __restrict__ dst,
                            const float* __restrict__ gw1,    // [518*4]
                            const float* __restrict__ econst, // gwsum[4], cb[4]
                            const float* __restrict__ w1_b,   // [4]
                            const float* __restrict__ ln2_g,  // [4]
                            const float* __restrict__ ln2_b,  // [4]
                            const float* __restrict__ w2_w,   // [4]
                            const float* __restrict__ w2_b,   // [1]
                            float* __restrict__ agg,
                            int n_edges) {
    __shared__ float s_gw1[EDIM * 4];
    for (int i = threadIdx.x; i < EDIM * 4; i += blockDim.x) s_gw1[i] = gw1[i];
    __syncthreads();

    const int lane = threadIdx.x & 31;
    const int wave = threadIdx.x >> 5;
    const int e = blockIdx.x * (blockDim.x >> 5) + wave;
    if (e >= n_edges) return;

    const int sn = src[e];
    const int dn = dst[e];

    float sx[8];
    float sum = 0.f, sq = 0.f, d0 = 0.f, d1 = 0.f, d2 = 0.f, d3 = 0.f;
#pragma unroll
    for (int j = 0; j < 8; ++j) {
        int c = lane + 32 * j;
        float v = mfeat[(size_t)sn * CDIM + c];
        sx[j] = v;
        sum += v; sq += v * v;
        const float* gr = &s_gw1[c * 4];
        d0 += v * gr[0]; d1 += v * gr[1]; d2 += v * gr[2]; d3 += v * gr[3];
        float u = mfeat[(size_t)dn * CDIM + c];
        sum += u; sq += u * u;
        const float* gr2 = &s_gw1[(CDIM + c) * 4];
        d0 += u * gr2[0]; d1 += u * gr2[1]; d2 += u * gr2[2]; d3 += u * gr2[3];
    }
    if (lane < 6) {   // positional extras: [pos_src(3), pos_dst(3)]
        float v = (lane < 3) ? x_pos[(size_t)sn * 3 + lane]
                             : x_pos[(size_t)dn * 3 + (lane - 3)];
        sum += v; sq += v * v;
        const float* gr = &s_gw1[(2 * CDIM + lane) * 4];
        d0 += v * gr[0]; d1 += v * gr[1]; d2 += v * gr[2]; d3 += v * gr[3];
    }
    // wave32 butterfly reduce (all lanes end with full sums)
#pragma unroll
    for (int off = 16; off > 0; off >>= 1) {
        sum += __shfl_xor(sum, off);
        sq  += __shfl_xor(sq,  off);
        d0  += __shfl_xor(d0,  off);
        d1  += __shfl_xor(d1,  off);
        d2  += __shfl_xor(d2,  off);
        d3  += __shfl_xor(d3,  off);
    }

    const float invN = 1.0f / (float)EDIM;
    float mu  = sum * invN;
    float var = sq * invN - mu * mu;
    float inv = rsqrtf(var + LN_EPS);

    float y[4];
    float dots[4] = {d0, d1, d2, d3};
#pragma unroll
    for (int k = 0; k < 4; ++k) {
        float yk = inv * (dots[k] - mu * econst[k]) + econst[4 + k] + w1_b[k];
        y[k] = gelu_exact(yk);
    }
    float mu2 = 0.25f * (y[0] + y[1] + y[2] + y[3]);
    float vr2 = 0.25f * (y[0]*y[0] + y[1]*y[1] + y[2]*y[2] + y[3]*y[3]) - mu2 * mu2;
    float inv2 = rsqrtf(vr2 + LN_EPS);
    float t = w2_b[0];
#pragma unroll
    for (int k = 0; k < 4; ++k)
        t += ((y[k] - mu2) * inv2 * ln2_g[k] + ln2_b[k]) * w2_w[k];
    float wgt = 1.0f / (1.0f + expf(-t));

#pragma unroll
    for (int j = 0; j < 8; ++j)
        atomicAdd(&agg[(size_t)dn * CDIM + lane + 32 * j], sx[j] * wgt);
}

// -------------------------------------------------------------------------
// Pooling + head
// -------------------------------------------------------------------------
__global__ void pool_kernel(const float* __restrict__ h,
                            const int* __restrict__ batch,
                            float* __restrict__ pooled,
                            float* __restrict__ cnt,
                            int n_nodes) {
    int idx = blockIdx.x * blockDim.x + threadIdx.x;
    if (idx >= n_nodes * CDIM) return;
    int node = idx >> 8, c = idx & (CDIM - 1);
    int b = batch[node];
    atomicAdd(&pooled[b * CDIM + c], h[idx]);
    if (c == 0) atomicAdd(&cnt[b], 1.0f);
}

__global__ void head_kernel(const float* __restrict__ pooled,
                            const float* __restrict__ cnt,
                            const float* __restrict__ head_w,
                            const float* __restrict__ head_b,
                            float* __restrict__ out) {
    int g = blockIdx.x, lane = threadIdx.x;
    float s = 0.f;
    for (int c = lane; c < CDIM; c += 32) s += pooled[g * CDIM + c] * head_w[c];
#pragma unroll
    for (int off = 16; off > 0; off >>= 1) s += __shfl_xor(s, off);
    if (lane == 0) out[g] = s / fmaxf(cnt[g], 1.0f) + head_b[0];
}

// -------------------------------------------------------------------------
extern "C" void kernel_launch(void* const* d_in, const int* in_sizes, int n_in,
                              void* d_out, int out_size, void* d_ws, size_t ws_size,
                              hipStream_t stream) {
    const float* x        = (const float*)d_in[0];
    const float* x_pos    = (const float*)d_in[1];
    const int*   eidx     = (const int*)  d_in[2];
    const int*   batch    = (const int*)  d_in[3];
    const float* dense_w  = (const float*)d_in[4];
    const float* dense_b  = (const float*)d_in[5];
    const float* d1_w     = (const float*)d_in[6];
    const float* d1_b     = (const float*)d_in[7];
    const float* ln1_g    = (const float*)d_in[8];
    const float* ln1_b    = (const float*)d_in[9];
    const float* w1_w     = (const float*)d_in[10];
    const float* w1_b     = (const float*)d_in[11];
    const float* ln2_g    = (const float*)d_in[12];
    const float* ln2_b    = (const float*)d_in[13];
    const float* w2_w     = (const float*)d_in[14];
    const float* w2_b     = (const float*)d_in[15];
    const float* d2_w     = (const float*)d_in[16];
    const float* d2_b     = (const float*)d_in[17];
    const float* head_w   = (const float*)d_in[18];
    const float* head_b   = (const float*)d_in[19];

    const int n_nodes = NNODES;
    const int n_edges = in_sizes[2] / 2;
    const int in_dim  = in_sizes[0] / n_nodes;        // 739
    const int in_kp   = (in_dim + 31) & ~31;          // 768
    const int L       = in_sizes[7] / CDIM;           // 4

    const int* src  = eidx;
    const int* dstp = eidx + n_edges;

    // ---- workspace layout (bytes) ----
    const size_t NC = (size_t)n_nodes * CDIM;         // 5.12M elems
    char* wsb = (char*)d_ws;
    float*  h       = (float*)wsb;                         wsb += NC * 4;       // f32
    __bf16* h_bf    = (__bf16*)wsb;                        wsb += NC * 2;       // bf16
    __bf16* agg_bf  = (__bf16*)wsb;                        wsb += NC * 2;       // bf16
    __bf16* wT      = (__bf16*)wsb;                        wsb += (size_t)CDIM * in_kp * 2; // max WT
    float*  gw1     = (float*)wsb;                         wsb += EDIM * 4 * 4;
    float*  econst  = (float*)wsb;                         wsb += 8 * 4;
    float*  pooled  = (float*)wsb;                         wsb += NGRAPHS * CDIM * 4;
    float*  cnt     = (float*)wsb;                         wsb += NGRAPHS * 4;
    // union region: x_bf (dead after first GEMM) overlaps m/agg
    char* ureg = wsb;
    __bf16* x_bf = (__bf16*)ureg;                          // n_nodes*in_kp bf16
    float*  mfe  = (float*)ureg;                           // NC f32
    float*  agg  = (float*)(ureg + NC * 4);                // NC f32

    const dim3 gblk(256);
    const dim3 ggrd((n_nodes + 127) / 128, CDIM / 64);
    const int nc = (int)NC;

    // ---- input packing ----
    {
        int n = n_nodes * in_kp;
        pack_cast_pad<<<(n + 255) / 256, 256, 0, stream>>>(x, x_bf, n_nodes, in_dim, in_kp);
        int nw = CDIM * in_kp;
        pack_transpose<<<(nw + 255) / 256, 256, 0, stream>>>(dense_w, wT, in_dim, CDIM, in_kp);
    }
    // h = x @ dense_w + dense_b   (writes f32 h and bf16 h_bf)
    gemm_wmma_kernel<<<ggrd, gblk, 0, stream>>>(x_bf, wT, dense_b, h, h_bf,
                                                n_nodes, CDIM, in_kp, /*act=*/0);

    const int nwt = CDIM * CDIM;
    for (int l = 0; l < L; ++l) {
        // m = gelu(h @ d1_w[l] + d1_b[l])
        pack_transpose<<<(nwt + 255) / 256, 256, 0, stream>>>(
            d1_w + (size_t)l * CDIM * CDIM, wT, CDIM, CDIM, CDIM);
        gemm_wmma_kernel<<<ggrd, gblk, 0, stream>>>(
            h_bf, wT, d1_b + (size_t)l * CDIM, mfe, (__bf16*)nullptr,
            n_nodes, CDIM, CDIM, /*act=*/1);
        // fold ln1 into w1
        prep_kernel<<<1, 256, 0, stream>>>(
            ln1_g + (size_t)l * EDIM, ln1_b + (size_t)l * EDIM,
            w1_w + (size_t)l * EDIM * 4, gw1, econst);
        // agg = 0
        zero_kernel<<<(nc + 255) / 256, 256, 0, stream>>>(agg, nc);
        // edge message + gated scatter-add
        edge_kernel<<<(n_edges + 7) / 8, 256, 0, stream>>>(
            mfe, x_pos, src, dstp, gw1, econst,
            w1_b + (size_t)l * 4, ln2_g + (size_t)l * 4, ln2_b + (size_t)l * 4,
            w2_w + (size_t)l * 4, w2_b + (size_t)l, agg, n_edges);
        // h = gelu(agg @ d2_w[l] + d2_b[l])
        pack_cast_pad<<<(nc + 255) / 256, 256, 0, stream>>>(agg, agg_bf, n_nodes, CDIM, CDIM);
        pack_transpose<<<(nwt + 255) / 256, 256, 0, stream>>>(
            d2_w + (size_t)l * CDIM * CDIM, wT, CDIM, CDIM, CDIM);
        gemm_wmma_kernel<<<ggrd, gblk, 0, stream>>>(
            agg_bf, wT, d2_b + (size_t)l * CDIM, h, h_bf,
            n_nodes, CDIM, CDIM, /*act=*/1);
    }

    // pooled/cnt = 0 (contiguous region: G*C + G)
    zero_kernel<<<(NGRAPHS * CDIM + NGRAPHS + 255) / 256, 256, 0, stream>>>(
        pooled, NGRAPHS * CDIM + NGRAPHS);
    pool_kernel<<<(nc + 255) / 256, 256, 0, stream>>>(h, batch, pooled, cnt, n_nodes);
    head_kernel<<<NGRAPHS, 32, 0, stream>>>(pooled, cnt, head_w, head_b, (float*)d_out);
}
